// MeshDecoder_39444979647132
// MI455X (gfx1250) — matrix operand order repaired and checked by reading, not compiled, not run on hardware
//
#include <hip/hip_runtime.h>

// ---------------------------------------------------------------------------
// MeshDecoder (SplineConv up-path) for gfx1250.
// Edge aggregation recast as [E, 9*Cin] x [9*Cin, Cout] GEMM executed with
// v_wmma_f32_16x16x32_bf16 (bf16 A/B, f32 accumulate), atomic scatter to dst.
// Templated on (Cin, Cout): A fragments and basis taps live in registers,
// the K loop is fully unrolled, only B-side weight tiles stream from LDS.
// ---------------------------------------------------------------------------

typedef __attribute__((ext_vector_type(16))) __bf16 v16bf;
typedef __attribute__((ext_vector_type(8)))  __bf16 v8bf;
typedef __attribute__((ext_vector_type(8)))  float  v8f;

union Frag {
  v16bf v;
  v8bf  h[2];
  uint4 q[2];
};
union Half8 { v8bf h; uint4 q; };
union B16   { __bf16 e[16]; uint4 q[2]; };

__device__ __forceinline__ unsigned short f2bf(float f) {
  unsigned u = __float_as_uint(f);
  u += 0x7FFFu + ((u >> 16) & 1u);          // round-to-nearest-even
  return (unsigned short)(u >> 16);
}

#define WPB 8   // waves per block (wave32)

// ---------------------------------------------------------------------------
// One wave per 16-edge tile:
//   agg[dst, :] += sum_k basis[e,k] * (x[src] @ W[k])
// A[e, k*Cin+c] = basis[e,k] * x[src[e], c]; W flat [9*Cin, Cout] is B.
// ---------------------------------------------------------------------------
template <int Cin, int Cout>
__global__ __launch_bounds__(256) void edge_spline_wmma(
    float* __restrict__ agg,
    const float* __restrict__ x,
    const float* __restrict__ Wg,
    const int* __restrict__ esrc,
    const int* __restrict__ edst,
    const float* __restrict__ pseudo,
    int E)
{
  constexpr int cs   = (Cin == 64) ? 6 : (Cin == 32) ? 5 : 4;  // log2(Cin)
  constexpr int Ktot = 9 * Cin;
  constexpr int nCh  = (Ktot + 31) / 32;
  constexpr int Kpad = nCh * 32;
  constexpr int nT   = Cout / 16;          // 16-wide column tiles (1 or 2)
  constexpr int NRUN = Cin / 16;           // distinct 8-ch runs per lane

  __shared__ unsigned short WT[Cout * Kpad];      // [Cout][Kpad] transposed bf16
  __shared__ unsigned int   xsU[WPB][16][Cin/2];  // packed bf16 pairs
  __shared__ unsigned short basisS[WPB][16][16];  // bf16 basis, zero-padded
  __shared__ int dstS[WPB][16];
  __shared__ int srcS[WPB][16];

  const int tid = threadIdx.x;

  // transpose W -> WT[o][K] (bf16), zero-pad K tail
  #pragma unroll 1
  for (int o = 0; o < Cout; ++o)
    for (int K = tid; K < Kpad; K += 256)
      WT[o * Kpad + K] = (K < Ktot) ? f2bf(Wg[K * Cout + o]) : (unsigned short)0;
  __syncthreads();

  const int wave = tid >> 5;
  const int lane = tid & 31;
  const int e0   = (blockIdx.x * WPB + wave) << 4;

  // lanes 0..15: edge metadata + 9-tap open-quadratic B-spline basis
  if (lane < 16) {
    const int e = e0 + lane;
    int sv = 0, dv = -1;
    float bb[9];
    if (e < E) {
      sv = esrc[e];
      dv = edst[e];
      const float tx = pseudo[2 * e + 0];
      const float ty = pseudo[2 * e + 1];
      const float bx0 = 0.5f * (1.f - tx) * (1.f - tx);
      const float bx1 = -tx * tx + tx + 0.5f;
      const float bx2 = 0.5f * tx * tx;
      const float by0 = 0.5f * (1.f - ty) * (1.f - ty);
      const float by1 = -ty * ty + ty + 0.5f;
      const float by2 = 0.5f * ty * ty;
      bb[0] = bx0 * by0; bb[1] = bx0 * by1; bb[2] = bx0 * by2;
      bb[3] = bx1 * by0; bb[4] = bx1 * by1; bb[5] = bx1 * by2;
      bb[6] = bx2 * by0; bb[7] = bx2 * by1; bb[8] = bx2 * by2;
    } else {
      #pragma unroll
      for (int k = 0; k < 9; ++k) bb[k] = 0.f;   // zero basis nullifies edge
    }
    srcS[wave][lane] = sv;
    dstS[wave][lane] = dv;
    #pragma unroll
    for (int k = 0; k < 9; ++k) basisS[wave][lane][k] = f2bf(bb[k]);
    #pragma unroll
    for (int k = 9; k < 16; ++k) basisS[wave][lane][k] = 0;
  }
  __builtin_amdgcn_wave_barrier();   // LDS is in-order within a wave

  // gather 16 rows of x[src] as packed bf16 pairs (branch-free)
  constexpr int halfC = Cin / 2;
  #pragma unroll
  for (int p = lane; p < 16 * halfC; p += 32) {
    const int m  = p >> (cs - 1);
    const int c2 = (p & (halfC - 1)) << 1;
    const float* xr = x + (size_t)srcS[wave][m] * Cin + c2;
    xsU[wave][m][c2 >> 1] =
        (unsigned)f2bf(xr[0]) | ((unsigned)f2bf(xr[1]) << 16);
  }
  __builtin_amdgcn_wave_barrier();

  const int mrow = lane & 15;
  const int hi   = (lane >> 4) & 1;
  const int kbA  = hi << 3;   // A K sub-base: 0 | 8   (16-bit A VGPR layout)
  const int kbB  = hi << 3;   // B K sub-base: 0 | 8   (16-bit B VGPR layout)
  const int ncol = lane & 15;

  // preload this lane's x-row fragments (<=4 b128 loads, then no A-side LDS)
  const unsigned int* xrow = &xsU[wave][mrow][0];
  Half8 xq[NRUN];
  #pragma unroll
  for (int i = 0; i < NRUN; ++i)
    xq[i].q = *(const uint4*)(xrow + (((kbA + 16 * i) & (Cin - 1)) >> 1));

  // preload basis row (16 bf16 in 2 b128 loads)
  B16 bfr;
  bfr.q[0] = *(const uint4*)(&basisS[wave][mrow][0]);
  bfr.q[1] = *(const uint4*)(&basisS[wave][mrow][8]);

  const unsigned short* w0 = &WT[(size_t)ncol * Kpad];
  const unsigned short* w1 = &WT[(size_t)(16 + ncol) * Kpad];

  v8f acc0 = {0.f, 0.f, 0.f, 0.f, 0.f, 0.f, 0.f, 0.f};
  v8f acc1 = {0.f, 0.f, 0.f, 0.f, 0.f, 0.f, 0.f, 0.f};

  #pragma unroll
  for (int kc = 0; kc < nCh; ++kc) {
    // tap index and run index are lane-independent & compile-time constants
    const int k0 = (kc * 32) >> cs;
    const int k1 = (kc * 32 + 16) >> cs;
    Frag a;
    a.h[0] = xq[(2 * kc) % NRUN].h * bfr.e[k0];       // v_pk_mul_bf16 x4
    a.h[1] = xq[(2 * kc + 1) % NRUN].h * bfr.e[k1];   // v_pk_mul_bf16 x4

    const int wb = kc * 32 + kbB;
    Frag b0;
    b0.q[0] = *(const uint4*)(w0 + wb);
    b0.q[1] = *(const uint4*)(w0 + wb + 16);
    acc0 = __builtin_amdgcn_wmma_f32_16x16x32_bf16(
        false, a.v, false, b0.v, (short)0, acc0, false, false);
    if constexpr (nT > 1) {
      Frag b1;
      b1.q[0] = *(const uint4*)(w1 + wb);
      b1.q[1] = *(const uint4*)(w1 + wb + 16);
      acc1 = __builtin_amdgcn_wmma_f32_16x16x32_bf16(
          false, a.v, false, b1.v, (short)0, acc1, false, false);
    }
  }

  // D layout: lanes 0-15 rows 0..7, lanes 16-31 rows 8..15; atomic scatter
  const int rbase = hi << 3;
  #pragma unroll
  for (int r = 0; r < 8; ++r) {
    const int d = dstS[wave][rbase + r];
    if (d >= 0) {
      atomicAdd(&agg[(size_t)d * Cout + ncol], acc0[r]);
      if constexpr (nT > 1)
        atomicAdd(&agg[(size_t)d * Cout + 16 + ncol], acc1[r]);
    }
  }
}

// agg[n,o] = bias[o] + x[n,:] @ R[:,o]   (root/self term; tiny FLOPs)
__global__ void node_init(float* __restrict__ agg,
                          const float* __restrict__ x,
                          const float* __restrict__ R,
                          const float* __restrict__ bias,
                          int N, int Cin, int Cout)
{
  const int i = blockIdx.x * blockDim.x + threadIdx.x;
  if (i >= N * Cout) return;
  const int n = i / Cout, o = i - n * Cout;
  float acc = bias[o];
  const float* xr = x + (size_t)n * Cin;
  for (int c = 0; c < Cin; ++c) acc += xr[c] * R[c * Cout + o];
  agg[i] = acc;
}

// out[n, 0:Cout] = relu(agg[n,:]); out[n, Cout:Cout+Cskip] = skip[n,:]
__global__ void post_concat(float* __restrict__ out,
                            const float* __restrict__ agg,
                            const float* __restrict__ skip,
                            int N, int Cout, int Cskip)
{
  const int Ct = Cout + Cskip;
  const int i = blockIdx.x * blockDim.x + threadIdx.x;
  if (i >= N * Ct) return;
  const int n = i / Ct, o = i - n * Ct;
  out[i] = (o < Cout) ? fmaxf(agg[(size_t)n * Cout + o], 0.f)
                      : skip[(size_t)n * Cskip + (o - Cout)];
}

// out[n,:] = in[idx[n],:]   (MeshUnpool)
__global__ void gather_rows(float* __restrict__ out,
                            const float* __restrict__ in,
                            const int* __restrict__ idx,
                            int N, int C)
{
  const int i = blockIdx.x * blockDim.x + threadIdx.x;
  if (i >= N * C) return;
  const int n = i / C, c = i - n * C;
  out[i] = in[(size_t)idx[n] * C + c];
}

static inline int cdiv(int a, int b) { return (a + b - 1) / b; }

static void launch_edge(float* agg, const float* x, const float* W,
                        const int* es, const int* ed, const float* ps,
                        int E, int Cin, int Cout, hipStream_t stream) {
  const int blocks = cdiv(E, 128);
  if (Cin == 64 && Cout == 32)
    edge_spline_wmma<64, 32><<<blocks, 256, 0, stream>>>(agg, x, W, es, ed, ps, E);
  else if (Cin == 32 && Cout == 32)
    edge_spline_wmma<32, 32><<<blocks, 256, 0, stream>>>(agg, x, W, es, ed, ps, E);
  else if (Cin == 32 && Cout == 16)
    edge_spline_wmma<32, 16><<<blocks, 256, 0, stream>>>(agg, x, W, es, ed, ps, E);
  else
    edge_spline_wmma<16, 16><<<blocks, 256, 0, stream>>>(agg, x, W, es, ed, ps, E);
}

extern "C" void kernel_launch(void* const* d_in, const int* in_sizes, int n_in,
                              void* d_out, int out_size, void* d_ws, size_t ws_size,
                              hipStream_t stream) {
  constexpr int N1 = 40000, N2 = 160000, E1 = 240000, E2 = 960000;
  constexpr int CI = 64, CM = 32, CO = 16;

  const float* x0      = (const float*)d_in[0];
  const int*   unpool1 = (const int*)  d_in[1];
  const int*   e1src   = (const int*)  d_in[2];
  const int*   e1dst   = e1src + E1;
  const float* pseudo1 = (const float*)d_in[3];
  const float* skip1   = (const float*)d_in[4];
  const int*   unpool2 = (const int*)  d_in[5];
  const int*   e2src   = (const int*)  d_in[6];
  const int*   e2dst   = e2src + E2;
  const float* pseudo2 = (const float*)d_in[7];
  const float* skip2   = (const float*)d_in[8];
  const float* W1a = (const float*)d_in[9],  *R1a = (const float*)d_in[10], *b1a = (const float*)d_in[11];
  const float* W2a = (const float*)d_in[12], *R2a = (const float*)d_in[13], *b2a = (const float*)d_in[14];
  const float* W1b = (const float*)d_in[15], *R1b = (const float*)d_in[16], *b1b = (const float*)d_in[17];
  const float* W2b = (const float*)d_in[18], *R2b = (const float*)d_in[19], *b2b = (const float*)d_in[20];

  // workspace layout (floats), with buffer reuse:
  float* ws   = (float*)d_ws;
  float* bufA = ws;                     // 2.56M : x1f -> h1b -> (A+B span = h2a)
  float* bufB = bufA + (size_t)N1 * CI; // 2.56M : h1a
  float* bufC = bufB + (size_t)N1 * CI; // 1.28M : h1c
  float* bufD = bufC + (size_t)N1 * CM; // 5.12M : x2f -> h2b
  float* aggB = bufD + (size_t)N2 * CM; // 2.56M : shared agg buffer

  // ---- level 1: unpool ----
  gather_rows<<<cdiv(N1 * CI, 256), 256, 0, stream>>>(bufA, x0, unpool1, N1, CI);

  // conv L1 #1 (W1a: 64->32), output concat skip1 -> h1a (64ch) in bufB
  node_init<<<cdiv(N1 * CM, 256), 256, 0, stream>>>(aggB, bufA, R1a, b1a, N1, CI, CM);
  launch_edge(aggB, bufA, W1a, e1src, e1dst, pseudo1, E1, CI, CM, stream);
  post_concat<<<cdiv(N1 * CI, 256), 256, 0, stream>>>(bufB, aggB, skip1, N1, CM, CM);

  // conv L1 #2 (W1a reused: 64->32) -> h1b (32ch) in bufA
  node_init<<<cdiv(N1 * CM, 256), 256, 0, stream>>>(aggB, bufB, R1a, b1a, N1, CI, CM);
  launch_edge(aggB, bufB, W1a, e1src, e1dst, pseudo1, E1, CI, CM, stream);
  post_concat<<<cdiv(N1 * CM, 256), 256, 0, stream>>>(bufA, aggB, nullptr, N1, CM, 0);

  // conv L1 #3 (W2a: 32->32) -> h1c (32ch) in bufC
  node_init<<<cdiv(N1 * CM, 256), 256, 0, stream>>>(aggB, bufA, R2a, b2a, N1, CM, CM);
  launch_edge(aggB, bufA, W2a, e1src, e1dst, pseudo1, E1, CM, CM, stream);
  post_concat<<<cdiv(N1 * CM, 256), 256, 0, stream>>>(bufC, aggB, nullptr, N1, CM, 0);

  // ---- level 2: unpool ----
  gather_rows<<<cdiv(N2 * CM, 256), 256, 0, stream>>>(bufD, bufC, unpool2, N2, CM);

  // conv L2 #1 (W1b: 32->16), concat skip2 -> h2a (32ch) in bufA(+bufB span)
  node_init<<<cdiv(N2 * CO, 256), 256, 0, stream>>>(aggB, bufD, R1b, b1b, N2, CM, CO);
  launch_edge(aggB, bufD, W1b, e2src, e2dst, pseudo2, E2, CM, CO, stream);
  post_concat<<<cdiv(N2 * CM, 256), 256, 0, stream>>>(bufA, aggB, skip2, N2, CO, CO);

  // conv L2 #2 (W1b reused: 32->16) -> h2b (16ch) in bufD
  node_init<<<cdiv(N2 * CO, 256), 256, 0, stream>>>(aggB, bufA, R1b, b1b, N2, CM, CO);
  launch_edge(aggB, bufA, W1b, e2src, e2dst, pseudo2, E2, CM, CO, stream);
  post_concat<<<cdiv(N2 * CO, 256), 256, 0, stream>>>(bufD, aggB, nullptr, N2, CO, 0);

  // conv L2 #3 (W2b: 16->16) -> d_out
  node_init<<<cdiv(N2 * CO, 256), 256, 0, stream>>>(aggB, bufD, R2b, b2b, N2, CO, CO);
  launch_edge(aggB, bufD, W2b, e2src, e2dst, pseudo2, E2, CO, CO, stream);
  post_concat<<<cdiv(N2 * CO, 256), 256, 0, stream>>>((float*)d_out, aggB, nullptr, N2, CO, 0);
}
